// BoundaryConsistencyLoss_31997506355846
// MI455X (gfx1250) — matrix operand order repaired and checked by reading, not compiled, not run on hardware
//
#include <hip/hip_runtime.h>
#include <hip/hip_bf16.h>

// BoundaryConsistencyLoss for MI455X (gfx1250, wave32).
// Window sums (w=5) computed as banded 16x16x32 f16 WMMA per 16-batch x 16-window tile.
// D = A(q-tile) x B(banded 0/1) + 0, f32 accumulate.

typedef __attribute__((ext_vector_type(16))) _Float16 v16h;
typedef __attribute__((ext_vector_type(8)))  _Float16 v8h;
typedef __attribute__((ext_vector_type(2)))  _Float16 v2h;
typedef __attribute__((ext_vector_type(8)))  float    v8f;

constexpr int BATCH = 512;
constexpr int LEN   = 16384;
constexpr int WIN   = 5;
constexpr int NW    = LEN - WIN + 1;   // 16380 windows
constexpr int TJ    = 128;             // windows per block (8 waves * 16)
constexpr int EST   = 144;             // staged elements per row (7*16 + 32)
constexpr int BT    = 16;              // batch rows per block

__global__ void bcl_zero(float* __restrict__ p, int n) {
  int i = blockIdx.x * blockDim.x + threadIdx.x;
  if (i < n) p[i] = 0.0f;
}

__global__ __launch_bounds__(256) void bcl_window_stats(
    const float* __restrict__ preds, const int* __restrict__ targets,
    const int* __restrict__ mask, float* __restrict__ mse_sum,
    float* __restrict__ msum_sum) {
  // f16 per-element quantities, row-major per batch row (t^2*m == t*m, so 4 suffice)
  __shared__ __align__(16) _Float16 s_m  [BT][EST];
  __shared__ __align__(16) _Float16 s_pm [BT][EST];
  __shared__ __align__(16) _Float16 s_p2m[BT][EST];
  __shared__ __align__(16) _Float16 s_tm [BT][EST];

  const int j0 = blockIdx.x * TJ;
  const int b0 = blockIdx.y * BT;
  const float4* __restrict__ preds4 = (const float4*)preds;   // 2 elements per float4
  const int2*   __restrict__ tg2    = (const int2*)targets;
  const int2*   __restrict__ mk2    = (const int2*)mask;

  // ---- Stage: global -> sigmoid -> f16 quantities in LDS ----
  // Two consecutive elements per thread: b128 preds + b64 targets + b64 mask.
  // (j0, LEN, EST all even -> pairs never straddle the L boundary, stores 4B aligned)
  for (int idx = threadIdx.x; idx < BT * (EST / 2); idx += 256) {
    const int row = idx / (EST / 2);
    const int ep  = (idx - row * (EST / 2)) * 2;  // even element offset
    const int l   = j0 + ep;
    float p0 = 0.f, p1 = 0.f, t0 = 0.f, t1 = 0.f, m0 = 0.f, m1 = 0.f;
    if (l < LEN) {
      const size_t off  = (size_t)(b0 + row) * LEN + l;  // even
      const size_t off2 = off >> 1;
      const float4 pr = preds4[off2];                    // [p0a,p1a,p0b,p1b]
      const int2   tt = tg2[off2];
      const int2   mm = mk2[off2];
      m0 = (float)mm.x; m1 = (float)mm.y;
      t0 = (float)tt.x; t1 = (float)tt.y;
      p0 = 1.0f / (1.0f + __expf(pr.x - pr.y));          // sigmoid(p1 - p0)
      p1 = 1.0f / (1.0f + __expf(pr.z - pr.w));
    }
    *(v2h*)&s_m  [row][ep] = (v2h){(_Float16)m0,          (_Float16)m1};
    *(v2h*)&s_pm [row][ep] = (v2h){(_Float16)(p0 * m0),   (_Float16)(p1 * m1)};
    *(v2h*)&s_p2m[row][ep] = (v2h){(_Float16)(p0 * p0 * m0), (_Float16)(p1 * p1 * m1)};
    *(v2h*)&s_tm [row][ep] = (v2h){(_Float16)(t0 * m0),   (_Float16)(t1 * m1)};
  }
  __syncthreads();

  // ---- Per-wave 16-window x 16-batch tile via banded WMMA ----
  const int lane  = threadIdx.x & 31;
  const int wave  = threadIdx.x >> 5;
  const int n     = lane & 15;                    // window within tile / D column
  const int hi8   = (lane & 16) ? 8 : 0;
  const int kbase = (lane & 16) ? 16 : 0;

  // B (32x16, f16): Band[k][n] = 1 iff n <= k <= n+4.  Per-lane half h -> k = hi8+h (h<8), 16+hi8+h-8 (h>=8).
  v16h bfrag;
#pragma unroll
  for (int h = 0; h < 16; ++h) {
    const int k = (h < 8) ? hi8 + h : 16 + hi8 + (h - 8);
    bfrag[h] = (k >= n && k <= n + 4) ? (_Float16)1.0f : (_Float16)0.0f;
  }

  // A (16x32, f16) lane layout: row M = lane&15; halves 0..7 -> K = hi8+h, halves 8..15 -> K = 16+hi8+h
  const int base = wave * 16;                     // element offset of this wave's tile
  const int off1 = base + hi8;                    // 8 contiguous halfs (16B aligned)
  const int off2 = base + 16 + hi8;

  auto loadA = [&](const _Float16 (&arr)[BT][EST]) -> v16h {
    const v8h r1 = *(const v8h*)&arr[n][off1];    // ds_load_b128
    const v8h r2 = *(const v8h*)&arr[n][off2];    // ds_load_b128
    v16h a;
#pragma unroll
    for (int h = 0; h < 8; ++h) { a[h] = r1[h]; a[h + 8] = r2[h]; }
    return a;
  };

  const v8f zero8 = {};
  v8f cm  = __builtin_amdgcn_wmma_f32_16x16x32_f16(false, loadA(s_m),   false, bfrag, (short)0, zero8, false, false);
  v8f cpm = __builtin_amdgcn_wmma_f32_16x16x32_f16(false, loadA(s_pm),  false, bfrag, (short)0, zero8, false, false);
  v8f cp2 = __builtin_amdgcn_wmma_f32_16x16x32_f16(false, loadA(s_p2m), false, bfrag, (short)0, zero8, false, false);
  v8f ctm = __builtin_amdgcn_wmma_f32_16x16x32_f16(false, loadA(s_tm),  false, bfrag, (short)0, zero8, false, false);

  // D layout: this lane's 8 values are window n for batch rows M = r (+8 if lane>=16)
  float accd2 = 0.0f, accms = 0.0f;
#pragma unroll
  for (int r = 0; r < 8; ++r) {
    const float msum = cm[r], spm = cpm[r], sp2m = cp2[r], stm = ctm[r];
    const float denom = fmaxf(msum, 1.0f);
    const float rd    = __builtin_amdgcn_rcpf(denom);  // denom >= 1 -> v_rcp_f32 is safe
    const float pmean = spm * rd;
    const float tmean = stm * rd;
    const float pvar  = (sp2m - 2.0f * pmean * spm + pmean * pmean * msum) * rd;
    const float tvar  = (stm  - 2.0f * tmean * stm + tmean * tmean * msum) * rd; // st2m == stm
    const float d = pvar - tvar;
    accd2 += d * d;
    accms += msum;
  }
  // fold batches 0..7 with 8..15 (lane ^ 16 holds same window n)
  accd2 += __shfl_xor(accd2, 16, 32);
  accms += __shfl_xor(accms, 16, 32);

  const int j = j0 + wave * 16 + n;
  if (lane < 16 && j < NW) {
    atomicAdd(&mse_sum[j],  accd2);
    atomicAdd(&msum_sum[j], accms);
  }
}

__global__ __launch_bounds__(256) void bcl_reduce(const float* __restrict__ mse_sum,
                                                  const float* __restrict__ msum_sum,
                                                  float* __restrict__ out) {
  __shared__ float s1[256];
  __shared__ float s2[256];
  float tot = 0.0f, cnt = 0.0f;
  for (int j = threadIdx.x; j < NW; j += 256) {
    const float valid = (msum_sum[j] > 0.0f) ? 1.0f : 0.0f;
    tot += mse_sum[j] * (1.0f / (float)BATCH) * valid;  // batch mean
    cnt += valid;
  }
  s1[threadIdx.x] = tot;
  s2[threadIdx.x] = cnt;
  __syncthreads();
  for (int s = 128; s > 0; s >>= 1) {
    if ((int)threadIdx.x < s) {
      s1[threadIdx.x] += s1[threadIdx.x + s];
      s2[threadIdx.x] += s2[threadIdx.x + s];
    }
    __syncthreads();
  }
  if (threadIdx.x == 0) out[0] = s1[0] / fmaxf(s2[0], 1.0f);  // CONSISTENCY_WEIGHT = 1.0
}

extern "C" void kernel_launch(void* const* d_in, const int* in_sizes, int n_in,
                              void* d_out, int out_size, void* d_ws, size_t ws_size,
                              hipStream_t stream) {
  (void)in_sizes; (void)n_in; (void)out_size; (void)ws_size;
  const float* preds   = (const float*)d_in[0];   // [512,16384,2] f32
  const int*   targets = (const int*)d_in[1];     // [512,16384] i32
  const int*   mask    = (const int*)d_in[2];     // [512,16384] i32

  float* mse_sum  = (float*)d_ws;                 // [NW]
  float* msum_sum = mse_sum + NW;                 // [NW]

  const int nzero = 2 * NW;
  bcl_zero<<<(nzero + 255) / 256, 256, 0, stream>>>(mse_sum, nzero);

  dim3 grid((NW + TJ - 1) / TJ, BATCH / BT);      // 128 x 32 blocks
  bcl_window_stats<<<grid, 256, 0, stream>>>(preds, targets, mask, mse_sum, msum_sum);

  bcl_reduce<<<1, 256, 0, stream>>>(mse_sum, msum_sum, (float*)d_out);
}
